// Encoder_42202348651194
// MI455X (gfx1250) — matrix operand order repaired and checked by reading, not compiled, not run on hardware
//
#include <hip/hip_runtime.h>
#include <hip/hip_bf16.h>

typedef __attribute__((ext_vector_type(16))) _Float16 v16h;
typedef __attribute__((ext_vector_type(8)))  _Float16 v8h;
typedef __attribute__((ext_vector_type(8)))  float    v8f;

#define BATCH  8192
#define NTOK   (BATCH * 5)
#define CMODEL 64
#define NHEAD  4
#define FFDIM  2048
#define NLAYER 2

// ---------------- WMMA helpers (CDNA5 16x16x32 f16 -> f32) ----------------
// A fragment (16x32 f16): lanes 0-15 hold row M=lane, halves [0..15]=K(half*16..half*16+15);
// lanes 16-31 same rows, other K-half.
__device__ __forceinline__ v16h load_a_frag(const _Float16* rowptr, int half) {
  v16h a;
  *(v8h*)&a         = *(const v8h*)(rowptr + half * 16);
  *((v8h*)&a + 1)   = *(const v8h*)(rowptr + half * 16 + 8);
  return a;
}

// B fragment (32x16 f16) from packed weights pk[kb][n][32]: lane r holds column
// n, K = half*16 .. half*16+15 -> 16 contiguous halves in the packed layout.
__device__ __forceinline__ v16h load_b_frag(const _Float16* __restrict__ pk,
                                            long kb, int N, int n, int half) {
  const _Float16* p = pk + (((kb * N) + n) << 5) + half * 16;
  v16h b;
  *(v8h*)&b       = *(const v8h*)(p);
  *((v8h*)&b + 1) = *(const v8h*)(p + 8);
  return b;
}

__device__ __forceinline__ v8f wmma32(v16h a, v16h b, v8f c) {
  return __builtin_amdgcn_wmma_f32_16x16x32_f16(false, a, false, b, (short)0, c,
                                                false, false);
}

// ---------------- weight packing: W[N,K] row-major f32 -> pk[Kpad/32][N][32] f16
__global__ void pack_kernel(const float* __restrict__ W, _Float16* __restrict__ pk,
                            int N, int K, int Kpad) {
  long tid = (long)blockIdx.x * blockDim.x + threadIdx.x;
  if (tid >= (long)N * Kpad) return;
  int n = (int)(tid / Kpad);
  int k = (int)(tid % Kpad);
  float v = (k < K) ? W[(long)n * K + k] : 0.0f;
  pk[(((long)(k >> 5) * N + n) << 5) + (k & 31)] = (_Float16)v;
}

// ---------------- feature extraction: one thread per batch item ----------------
__global__ void feats_kernel(const int* __restrict__ ai, const int* __restrict__ aj,
                             const int* __restrict__ gi, const int* __restrict__ gj,
                             const int* __restrict__ walls, _Float16* __restrict__ F) {
  int b = blockIdx.x * blockDim.x + threadIdx.x;
  if (b >= BATCH) return;
  const int* wb = walls + (long)b * 256;
  int w0 = -1, w1 = -1, w2 = -1;
  for (int i = 0; i < 256; ++i) {
    if (wb[i] == 1) {
      if (w0 < 0) w0 = i;
      else if (w1 < 0) w1 = i;
      else { w2 = i; break; }
    }
  }
  int   widx[3] = {w0, w1, w2};
  float valid[3], win[3], wjn[3];
  #pragma unroll
  for (int k = 0; k < 3; ++k) {
    int s = widx[k] < 0 ? 0 : widx[k];
    valid[k] = widx[k] < 0 ? 0.0f : 1.0f;
    win[k] = 2.0f * (float)(s >> 4) / 15.0f - 1.0f;
    wjn[k] = 2.0f * (float)(s & 15) / 15.0f - 1.0f;
  }
  int aib = ai[b], ajb = aj[b];
  float gin = 2.0f * (float)gi[b] / 15.0f - 1.0f;
  float gjn = 2.0f * (float)gj[b] / 15.0f - 1.0f;
  const int mvi[5] = {0, -1, 1, 0, 0};
  const int mvj[5] = {0, 0, 0, -1, 1};
  for (int p = 0; p < 5; ++p) {
    int ni = aib + mvi[p], nj = ajb + mvj[p];
    bool inb = (ni >= 0) && (ni < 16) && (nj >= 0) && (nj < 16);
    int pi = inb ? ni : aib, pj = inb ? nj : ajb;
    float f[55];
    float ain = 2.0f * (float)pi / 15.0f - 1.0f;
    float ajn = 2.0f * (float)pj / 15.0f - 1.0f;
    float dxg = gjn - ajn, dyg = gin - ain;
    float dg = sqrtf(dxg * dxg + dyg * dyg + 1e-8f);
    f[0] = ain; f[1] = ajn; f[2] = gin; f[3] = gjn;
    f[4] = dxg; f[5] = dyg; f[6] = dg; f[7] = dyg / dg; f[8] = dxg / dg;
    #pragma unroll
    for (int k = 0; k < 3; ++k) {          // walls relative to agent pos
      float dx = wjn[k] - ajn, dy = win[k] - ain;
      float d = sqrtf(dx * dx + dy * dy + 1e-8f);
      float* o = f + 9 + k * 7;
      o[0] = win[k] * valid[k]; o[1] = wjn[k] * valid[k];
      o[2] = dx * valid[k];     o[3] = dy * valid[k];
      o[4] = d * valid[k];      o[5] = (dy / d) * valid[k]; o[6] = (dx / d) * valid[k];
    }
    #pragma unroll
    for (int k = 0; k < 3; ++k) {          // walls relative to goal
      float dx = wjn[k] - gjn, dy = win[k] - gin;
      float d = sqrtf(dx * dx + dy * dy + 1e-8f);
      float* o = f + 30 + k * 7;
      o[0] = win[k] * valid[k]; o[1] = wjn[k] * valid[k];
      o[2] = dx * valid[k];     o[3] = dy * valid[k];
      o[4] = d * valid[k];      o[5] = (dy / d) * valid[k]; o[6] = (dx / d) * valid[k];
    }
    f[51] = (pi > 0)  ? (float)wb[(pi - 1) * 16 + pj] : 1.0f;
    f[52] = (pi < 15) ? (float)wb[(pi + 1) * 16 + pj] : 1.0f;
    f[53] = (pj > 0)  ? (float)wb[pi * 16 + pj - 1]   : 1.0f;
    f[54] = (pj < 15) ? (float)wb[pi * 16 + pj + 1]   : 1.0f;
    _Float16* out = F + (long)(b * 5 + p) * 64;
    for (int i = 0; i < 55; ++i) out[i] = (_Float16)f[i];
    for (int i = 55; i < 64; ++i) out[i] = (_Float16)0.0f;
  }
}

// ---------------- generic WMMA GEMM: C[M,N] = act(A[M,K] @ Wt + bias) --------
// block = 256 threads = 8 waves; block tile 256(M) x 64(N); wave tile 32 x 64.
// Two row-groups per wave share every B fragment -> 1.5 b128 loads per WMMA.
__global__ __launch_bounds__(256) void gemm_kernel(
    const _Float16* __restrict__ A, const _Float16* __restrict__ PK,
    const float* __restrict__ bias, float* __restrict__ Cf,
    _Float16* __restrict__ Ch, int K, int N, int act) {
  const int wave = threadIdx.x >> 5, lane = threadIdx.x & 31;
  const int r = lane & 15, half = lane >> 4;
  const int m0 = blockIdx.x * 256 + wave * 32;
  const int n0 = blockIdx.y * 64;
  v8f acc[2][4] = {};
  const int KB = K >> 5;
  for (int kb = 0; kb < KB; ++kb) {
    v16h af0 = load_a_frag(A + (long)(m0 + r) * K + kb * 32, half);
    v16h af1 = load_a_frag(A + (long)(m0 + 16 + r) * K + kb * 32, half);
    #pragma unroll
    for (int t = 0; t < 4; ++t) {
      v16h bf = load_b_frag(PK, kb, N, n0 + t * 16 + r, half);
      acc[0][t] = wmma32(af0, bf, acc[0][t]);
      acc[1][t] = wmma32(af1, bf, acc[1][t]);
    }
  }
  #pragma unroll
  for (int g = 0; g < 2; ++g) {
    #pragma unroll
    for (int t = 0; t < 4; ++t) {
      int col = n0 + t * 16 + r;
      float bb = bias[col];
      #pragma unroll
      for (int v = 0; v < 8; ++v) {
        int row = m0 + g * 16 + v + half * 8;
        float x = acc[g][t][v] + bb;
        if (act == 1)      x = 0.5f * x * (1.0f + erff(x * 0.70710678118f)); // exact GELU
        else if (act == 2) x = fmaxf(x, 0.0f);
        long o = (long)row * N + col;
        if (Cf) Cf[o] = x;
        if (Ch) Ch[o] = (_Float16)x;
      }
    }
  }
}

// ---------------- fused FFN: Y = relu(X @ W1^T + b1) @ W2^T + b2 -------------
// wave tile 32 rows; FF processed in 64-wide chunks; hidden tile bounces
// through a private per-wave LDS region (same-wave DS ordering, no barriers).
__global__ __launch_bounds__(256) void ffn_kernel(
    const _Float16* __restrict__ Xh, const _Float16* __restrict__ PK1,
    const float* __restrict__ b1, const _Float16* __restrict__ PK2,
    const float* __restrict__ b2, float* __restrict__ Y) {
  __shared__ __align__(16) _Float16 hs[8][32 * 64];
  const int wave = threadIdx.x >> 5, lane = threadIdx.x & 31;
  const int r = lane & 15, half = lane >> 4;
  const int m0 = blockIdx.x * 256 + wave * 32;
  v16h af1[2][2];
  #pragma unroll
  for (int g = 0; g < 2; ++g) {
    af1[g][0] = load_a_frag(Xh + (long)(m0 + g * 16 + r) * 64, half);
    af1[g][1] = load_a_frag(Xh + (long)(m0 + g * 16 + r) * 64 + 32, half);
  }
  v8f acc2[2][4] = {};
  for (int ch = 0; ch < FFDIM / 64; ++ch) {
    v8f hacc[2][4] = {};
    #pragma unroll
    for (int kb = 0; kb < 2; ++kb) {
      #pragma unroll
      for (int t = 0; t < 4; ++t) {
        v16h bf = load_b_frag(PK1, kb, FFDIM, ch * 64 + t * 16 + r, half);
        hacc[0][t] = wmma32(af1[0][kb], bf, hacc[0][t]);
        hacc[1][t] = wmma32(af1[1][kb], bf, hacc[1][t]);
      }
    }
    // bias + relu, stash 32x64 hidden tile in this wave's LDS region (f16)
    #pragma unroll
    for (int g = 0; g < 2; ++g) {
      #pragma unroll
      for (int t = 0; t < 4; ++t) {
        float bb = b1[ch * 64 + t * 16 + r];
        #pragma unroll
        for (int v = 0; v < 8; ++v) {
          float x = fmaxf(hacc[g][t][v] + bb, 0.0f);
          hs[wave][(g * 16 + v + half * 8) * 64 + t * 16 + r] = (_Float16)x;
        }
      }
    }
    // second GEMM stage: hidden(32x64) as A operand, accumulate into 32x64
    #pragma unroll
    for (int ks = 0; ks < 2; ++ks) {
      v16h af2_0 = load_a_frag(&hs[wave][(r)      * 64 + ks * 32], half);
      v16h af2_1 = load_a_frag(&hs[wave][(16 + r) * 64 + ks * 32], half);
      long kb2 = (long)ch * 2 + ks;
      #pragma unroll
      for (int t = 0; t < 4; ++t) {
        v16h bf = load_b_frag(PK2, kb2, 64, t * 16 + r, half);
        acc2[0][t] = wmma32(af2_0, bf, acc2[0][t]);
        acc2[1][t] = wmma32(af2_1, bf, acc2[1][t]);
      }
    }
  }
  #pragma unroll
  for (int g = 0; g < 2; ++g) {
    #pragma unroll
    for (int t = 0; t < 4; ++t) {
      float bb = b2[t * 16 + r];
      #pragma unroll
      for (int v = 0; v < 8; ++v)
        Y[(long)(m0 + g * 16 + v + half * 8) * 64 + t * 16 + r] = acc2[g][t][v] + bb;
    }
  }
}

// ---------------- tiny 5x5 attention, one thread per (b, head, query) --------
__global__ void attn_kernel(const float* __restrict__ qkv, _Float16* __restrict__ O) {
  int tid = blockIdx.x * blockDim.x + threadIdx.x;
  if (tid >= BATCH * NHEAD * 5) return;
  int q = tid % 5, h = (tid / 5) % NHEAD, b = tid / (5 * NHEAD);
  long base = (long)b * 5;
  const float* qp = qkv + (base + q) * 192 + h * 16;
  float s[5];
  #pragma unroll
  for (int k = 0; k < 5; ++k) {
    const float* kp = qkv + (base + k) * 192 + 64 + h * 16;
    float d = 0.0f;
    #pragma unroll
    for (int e = 0; e < 16; ++e) d += qp[e] * kp[e];
    s[k] = d * 0.25f;   // 1/sqrt(16)
  }
  float m = s[0];
  #pragma unroll
  for (int k = 1; k < 5; ++k) m = fmaxf(m, s[k]);
  float e[5], sum = 0.0f;
  #pragma unroll
  for (int k = 0; k < 5; ++k) { e[k] = expf(s[k] - m); sum += e[k]; }
  float inv = 1.0f / sum;
  float o[16] = {};
  #pragma unroll
  for (int k = 0; k < 5; ++k) {
    float w = e[k] * inv;
    const float* vp = qkv + (base + k) * 192 + 128 + h * 16;
    #pragma unroll
    for (int d = 0; d < 16; ++d) o[d] += w * vp[d];
  }
  _Float16* op = O + (base + q) * 64 + h * 16;
  #pragma unroll
  for (int d = 0; d < 16; ++d) op[d] = (_Float16)o[d];
}

// ---------------- residual + LayerNorm, one wave per token -------------------
__global__ __launch_bounds__(256) void ln_kernel(
    const float* __restrict__ xin, const float* __restrict__ dlt,
    const float* __restrict__ g, const float* __restrict__ bt,
    float* __restrict__ xout, _Float16* __restrict__ xh) {
  int wave = threadIdx.x >> 5, lane = threadIdx.x & 31;
  long t = (long)blockIdx.x * 8 + wave;
  const float* xi = xin + t * 64;
  const float* dl = dlt + t * 64;
  float a0 = xi[lane] + dl[lane];
  float a1 = xi[lane + 32] + dl[lane + 32];
  float sum = a0 + a1;
  #pragma unroll
  for (int o = 16; o > 0; o >>= 1) sum += __shfl_xor(sum, o, 32);
  float mean = sum * (1.0f / 64.0f);
  float d0 = a0 - mean, d1 = a1 - mean;
  float vs = d0 * d0 + d1 * d1;
  #pragma unroll
  for (int o = 16; o > 0; o >>= 1) vs += __shfl_xor(vs, o, 32);
  float inv = rsqrtf(vs * (1.0f / 64.0f) + 1e-5f);
  float y0 = d0 * inv * g[lane] + bt[lane];
  float y1 = d1 * inv * g[lane + 32] + bt[lane + 32];
  xout[t * 64 + lane] = y0;
  xout[t * 64 + lane + 32] = y1;
  xh[t * 64 + lane] = (_Float16)y0;
  xh[t * 64 + lane + 32] = (_Float16)y1;
}

// -----------------------------------------------------------------------------
extern "C" void kernel_launch(void* const* d_in, const int* in_sizes, int n_in,
                              void* d_out, int out_size, void* d_ws, size_t ws_size,
                              hipStream_t stream) {
  (void)in_sizes; (void)n_in; (void)out_size; (void)ws_size;
  const int*   ai        = (const int*)d_in[0];
  const int*   aj        = (const int*)d_in[1];
  const int*   gi        = (const int*)d_in[2];
  const int*   gj        = (const int*)d_in[3];
  const int*   walls     = (const int*)d_in[4];
  const float* mlp_w1    = (const float*)d_in[5];
  const float* mlp_b1    = (const float*)d_in[6];
  const float* mlp_w2    = (const float*)d_in[7];
  const float* mlp_b2    = (const float*)d_in[8];
  const float* in_proj_w = (const float*)d_in[9];
  const float* in_proj_b = (const float*)d_in[10];
  const float* out_w     = (const float*)d_in[11];
  const float* out_b     = (const float*)d_in[12];
  const float* ln1_g     = (const float*)d_in[13];
  const float* ln1_b     = (const float*)d_in[14];
  const float* ln2_g     = (const float*)d_in[15];
  const float* ln2_b     = (const float*)d_in[16];
  const float* ffn_w1    = (const float*)d_in[17];
  const float* ffn_b1    = (const float*)d_in[18];
  const float* ffn_w2    = (const float*)d_in[19];
  const float* ffn_b2    = (const float*)d_in[20];
  float* out = (float*)d_out;

  // -------- workspace carve-out --------
  char* ws = (char*)d_ws;
  size_t cur = 0;
  auto alloc = [&](size_t bytes) -> void* {
    void* p = ws + cur;
    cur += (bytes + 255) & ~(size_t)255;
    return p;
  };
  _Float16* pk_mlp1 = (_Float16*)alloc((size_t)2 * 128 * 32 * 2);
  _Float16* pk_mlp2 = (_Float16*)alloc((size_t)4 * 64 * 32 * 2);
  _Float16* pk_qkv  = (_Float16*)alloc((size_t)NLAYER * 2 * 192 * 32 * 2);
  _Float16* pk_out  = (_Float16*)alloc((size_t)NLAYER * 2 * 64 * 32 * 2);
  _Float16* pk_f1   = (_Float16*)alloc((size_t)NLAYER * 2 * FFDIM * 32 * 2);
  _Float16* pk_f2   = (_Float16*)alloc((size_t)NLAYER * 64 * 64 * 32 * 2);
  _Float16* feats   = (_Float16*)alloc((size_t)NTOK * 64 * 2);
  _Float16* h1      = (_Float16*)alloc((size_t)NTOK * 128 * 2);
  float*    x       = (float*)   alloc((size_t)NTOK * 64 * 4);
  _Float16* xh      = (_Float16*)alloc((size_t)NTOK * 64 * 2);
  float*    qkv     = (float*)   alloc((size_t)NTOK * 192 * 4);
  _Float16* attnh   = (_Float16*)alloc((size_t)NTOK * 64 * 2);
  float*    y       = (float*)   alloc((size_t)NTOK * 64 * 4);

  auto pack = [&](const float* W, _Float16* pk, int N, int K, int Kpad) {
    long tot = (long)N * Kpad;
    pack_kernel<<<(int)((tot + 255) / 256), 256, 0, stream>>>(W, pk, N, K, Kpad);
  };

  // -------- pack all weights into WMMA-B-friendly f16 layout --------
  pack(mlp_w1, pk_mlp1, 128, 55, 64);
  pack(mlp_w2, pk_mlp2, 64, 128, 128);
  for (int l = 0; l < NLAYER; ++l) {
    pack(in_proj_w + (long)l * 192 * 64, pk_qkv + (long)l * 2 * 192 * 32, 192, 64, 64);
    pack(out_w     + (long)l * 64 * 64,  pk_out + (long)l * 2 * 64 * 32,   64, 64, 64);
    pack(ffn_w1    + (long)l * FFDIM * 64, pk_f1 + (long)l * 2 * FFDIM * 32, FFDIM, 64, 64);
    pack(ffn_w2    + (long)l * 64 * FFDIM, pk_f2 + (long)l * 64 * 64 * 32, 64, FFDIM, FFDIM);
  }

  // -------- features (f16, K padded 55 -> 64 with zeros) --------
  feats_kernel<<<BATCH / 256, 256, 0, stream>>>(ai, aj, gi, gj, walls, feats);

  const int MB = NTOK / 256;  // 160 M-blocks (256 rows each)

  // -------- feature MLP: Linear(55->128)+GELU, Linear(128->64) --------
  gemm_kernel<<<dim3(MB, 2), 256, 0, stream>>>(feats, pk_mlp1, mlp_b1, nullptr, h1, 64, 128, 1);
  gemm_kernel<<<dim3(MB, 1), 256, 0, stream>>>(h1, pk_mlp2, mlp_b2, x, xh, 128, 64, 0);

  // -------- transformer layers --------
  for (int l = 0; l < NLAYER; ++l) {
    gemm_kernel<<<dim3(MB, 3), 256, 0, stream>>>(
        xh, pk_qkv + (long)l * 2 * 192 * 32, in_proj_b + (long)l * 192,
        qkv, nullptr, 64, 192, 0);
    attn_kernel<<<(BATCH * NHEAD * 5) / 256, 256, 0, stream>>>(qkv, attnh);
    gemm_kernel<<<dim3(MB, 1), 256, 0, stream>>>(
        attnh, pk_out + (long)l * 2 * 64 * 32, out_b + (long)l * 64,
        y, nullptr, 64, 64, 0);
    ln_kernel<<<NTOK / 8, 256, 0, stream>>>(x, y, ln1_g + (long)l * 64,
                                            ln1_b + (long)l * 64, x, xh);
    ffn_kernel<<<MB, 256, 0, stream>>>(
        xh, pk_f1 + (long)l * 2 * FFDIM * 32, ffn_b1 + (long)l * FFDIM,
        pk_f2 + (long)l * 64 * 64 * 32, ffn_b2 + (long)l * 64, y);
    float* xdst = (l == NLAYER - 1) ? out : x;
    ln_kernel<<<NTOK / 8, 256, 0, stream>>>(x, y, ln2_g + (long)l * 64,
                                            ln2_b + (long)l * 64, xdst, xh);
  }
}
